// GroupAttention_19164144075401
// MI455X (gfx1250) — compile-verified, matrix operand order checked by reading, not run on hardware
//
#include <hip/hip_runtime.h>

// ---------------- constants ----------------
constexpr int kB = 4;
constexpr int kS = 2048;
constexpr int kD = 1024;
constexpr long long kSS = (long long)kS * kS;   // 4194304 = 2^22
constexpr int kLdsStride = 40;                  // halfwords; keeps b128 stores 16B-aligned

// ---------------- WMMA types ----------------
typedef __attribute__((ext_vector_type(16))) __bf16 v16bf;
typedef __attribute__((ext_vector_type(8)))  float  v8f;

union FragBF { v16bf v; unsigned u[8]; };

// fp32 -> bf16 round-to-nearest-even (bit twiddling; avoids __bf16 arithmetic)
static __device__ __forceinline__ unsigned short f2bf(float f) {
    unsigned u = __float_as_uint(f);
    unsigned r = u + 0x7fffu + ((u >> 16) & 1u);
    return (unsigned short)(r >> 16);
}

// Fragment gather: A and (pre-transposed) B both live in LDS as
// [row][k] with row-contiguous K, stride kLdsStride halfwords.
// 16-bit A-matrix layout: lanes 0-15 = M rows, half selects K-octet,
// VGPR r holds K pair ((r&4)?16:0) + half*8 + (r&3)*2.
static __device__ __forceinline__ FragBF load_frag(const unsigned short* lds,
                                                   int rowBase, int l15,
                                                   const int* kks) {
    FragBF f;
#pragma unroll
    for (int r = 0; r < 8; ++r)
        f.u[r] = *(const unsigned*)&lds[(rowBase + l15) * kLdsStride + kks[r]];
    return f;
}

// =====================================================================
// Kernel 1: LayerNorm over D=1024, output bf16.  One block per row.
// =====================================================================
__global__ __launch_bounds__(256) void ln_kernel(const float* __restrict__ ctx,
                                                 const float* __restrict__ gamma,
                                                 const float* __restrict__ beta,
                                                 unsigned short* __restrict__ xbf) {
    const int row = blockIdx.x;        // 0..B*S-1
    const int tid = threadIdx.x;
    const float4 x = ((const float4*)(ctx + (size_t)row * kD))[tid];

    float s  = x.x + x.y + x.z + x.w;
    float ss = x.x*x.x + x.y*x.y + x.z*x.z + x.w*x.w;

    __shared__ float r1[256], r2[256];
    r1[tid] = s; r2[tid] = ss;
    __syncthreads();
    for (int off = 128; off > 0; off >>= 1) {
        if (tid < off) { r1[tid] += r1[tid + off]; r2[tid] += r2[tid + off]; }
        __syncthreads();
    }
    const float mean = r1[0] * (1.0f / kD);
    const float var  = r2[0] * (1.0f / kD) - mean * mean;
    const float rstd = rsqrtf(var + 1e-3f);

    const float4 g  = ((const float4*)gamma)[tid];
    const float4 be = ((const float4*)beta)[tid];
    const float y0 = (x.x - mean) * rstd * g.x + be.x;
    const float y1 = (x.y - mean) * rstd * g.y + be.y;
    const float y2 = (x.z - mean) * rstd * g.z + be.z;
    const float y3 = (x.w - mean) * rstd * g.w + be.w;

    uint2 pk;
    pk.x = (unsigned)f2bf(y0) | ((unsigned)f2bf(y1) << 16);
    pk.y = (unsigned)f2bf(y2) | ((unsigned)f2bf(y3) << 16);
    ((uint2*)(xbf + (size_t)row * kD))[tid] = pk;
}

// =====================================================================
// Kernel 2: transpose + fp32->bf16 for weights: Wt[n][k] = W[k][n].
// Both GEMMs can then stage B tiles with contiguous-K b128 copies.
// 64x64 tile per block via LDS.
// =====================================================================
__global__ __launch_bounds__(256) void transposebf_kernel(const float* __restrict__ w,
                                                          unsigned short* __restrict__ wt) {
    __shared__ unsigned short t[64 * 65];
    const int tid  = threadIdx.x;
    const int rB   = blockIdx.x * 64;      // source row base (k)
    const int cB   = blockIdx.y * 64;      // source col base (n)
    const int r0   = tid >> 4;             // 0..15
    const int c4   = (tid & 15) * 4;       // 0..60

#pragma unroll
    for (int it = 0; it < 4; ++it) {
        const int row = r0 + it * 16;
        const float4 v = *(const float4*)(w + (size_t)(rB + row) * kD + cB + c4);
        t[(c4 + 0) * 65 + row] = f2bf(v.x);
        t[(c4 + 1) * 65 + row] = f2bf(v.y);
        t[(c4 + 2) * 65 + row] = f2bf(v.z);
        t[(c4 + 3) * 65 + row] = f2bf(v.w);
    }
    __syncthreads();
#pragma unroll
    for (int it = 0; it < 4; ++it) {
        const int row = r0 + it * 16;      // output row (= source col)
        uint2 pk;
        pk.x = (unsigned)t[row * 65 + c4]     | ((unsigned)t[row * 65 + c4 + 1] << 16);
        pk.y = (unsigned)t[row * 65 + c4 + 2] | ((unsigned)t[row * 65 + c4 + 3] << 16);
        *(uint2*)(wt + (size_t)(cB + row) * kD + rB + c4) = pk;
    }
}

// =====================================================================
// Shared GEMM core: C[64x128 block] = A[64xK] * B^T[128xK]  (both bf16,
// both row-major with contiguous K), f32 accumulate via WMMA.
// 8 waves in 2(M) x 4(N) grid; each wave owns a 32x32 tile = 4 WMMAs/step.
// =====================================================================
struct GemmAcc { v8f a[2][2]; };

static __device__ __forceinline__ void gemm_core(const unsigned short* __restrict__ aG,
                                                 const unsigned short* __restrict__ bG,
                                                 int mBase, int nBase,
                                                 unsigned short* ldsA,
                                                 unsigned short* ldsB,
                                                 GemmAcc& acc,
                                                 int wm, int wn, int half, int l15) {
    const int tid = threadIdx.x;
    // staging coordinates (fixed trip counts -> clean b128 copies)
    const int arow = tid >> 2;             // 0..63
    const int ak8  = (tid & 3) * 8;        // 0,8,16,24
    const int brow0 = tid >> 2;            // 0..63   (B rows 0..127 in 2 steps)
    const int bk8   = (tid & 3) * 8;

    int kks[8];
#pragma unroll
    for (int r = 0; r < 8; ++r) kks[r] = ((r & 4) ? 16 : 0) + half * 8 + (r & 3) * 2;

    for (int k0 = 0; k0 < kD; k0 += 32) {
        // A tile: 64 rows x 32 k
        *(uint4*)&ldsA[arow * kLdsStride + ak8] =
            *(const uint4*)(aG + (size_t)(mBase + arow) * kD + k0 + ak8);
        // B tile: 128 rows x 32 k
        *(uint4*)&ldsB[brow0 * kLdsStride + bk8] =
            *(const uint4*)(bG + (size_t)(nBase + brow0) * kD + k0 + bk8);
        *(uint4*)&ldsB[(brow0 + 64) * kLdsStride + bk8] =
            *(const uint4*)(bG + (size_t)(nBase + brow0 + 64) * kD + k0 + bk8);
        if (k0 + 32 < kD) {   // prefetch next K tile -> global_prefetch_b8
            __builtin_prefetch(aG + (size_t)(mBase + arow) * kD + k0 + 32 + ak8, 0, 1);
            __builtin_prefetch(bG + (size_t)(nBase + brow0) * kD + k0 + 32 + bk8, 0, 1);
            __builtin_prefetch(bG + (size_t)(nBase + brow0 + 64) * kD + k0 + 32 + bk8, 0, 1);
        }
        __syncthreads();

        const FragBF a0 = load_frag(ldsA, wm * 32,      l15, kks);
        const FragBF a1 = load_frag(ldsA, wm * 32 + 16, l15, kks);
        const FragBF b0 = load_frag(ldsB, wn * 32,      l15, kks);
        const FragBF b1 = load_frag(ldsB, wn * 32 + 16, l15, kks);

        acc.a[0][0] = __builtin_amdgcn_wmma_f32_16x16x32_bf16(false, a0.v, false, b0.v,
                                                              (short)0, acc.a[0][0], false, false);
        acc.a[0][1] = __builtin_amdgcn_wmma_f32_16x16x32_bf16(false, a0.v, false, b1.v,
                                                              (short)0, acc.a[0][1], false, false);
        acc.a[1][0] = __builtin_amdgcn_wmma_f32_16x16x32_bf16(false, a1.v, false, b0.v,
                                                              (short)0, acc.a[1][0], false, false);
        acc.a[1][1] = __builtin_amdgcn_wmma_f32_16x16x32_bf16(false, a1.v, false, b1.v,
                                                              (short)0, acc.a[1][1], false, false);
        __syncthreads();
    }
}

// =====================================================================
// Kernel 3: projection GEMM  out[8192,1024] = Xbf @ Wt^T + bias, bf16 out
// =====================================================================
__global__ __launch_bounds__(256) void gemm_proj_kernel(const unsigned short* __restrict__ xbf,
                                                        const unsigned short* __restrict__ wtbf,
                                                        const float* __restrict__ bias,
                                                        unsigned short* __restrict__ outbf) {
    __shared__ __align__(16) unsigned short ldsA[64 * kLdsStride];
    __shared__ __align__(16) unsigned short ldsB[128 * kLdsStride];

    const int tid  = threadIdx.x;
    const int wave = tid >> 5, lane = tid & 31;
    const int wm = wave >> 2, wn = wave & 3;
    const int half = lane >> 4, l15 = lane & 15;
    const int mBase = blockIdx.x * 64;
    const int nBase = blockIdx.y * 128;

    GemmAcc acc = {};
    gemm_core(xbf, wtbf, mBase, nBase, ldsA, ldsB, acc, wm, wn, half, l15);

#pragma unroll
    for (int ti = 0; ti < 2; ++ti)
#pragma unroll
        for (int tj = 0; tj < 2; ++tj)
#pragma unroll
            for (int r = 0; r < 8; ++r) {
                const int gm = mBase + wm * 32 + ti * 16 + half * 8 + r;
                const int gn = nBase + wn * 32 + tj * 16 + l15;
                outbf[(size_t)gm * kD + gn] = f2bf(acc.a[ti][tj][r] + bias[gn]);
            }
}

// =====================================================================
// Kernel 4: scores[b,i,j] = q[b,i,:]·k[b,j,:]/D + mask, fp32 out
// =====================================================================
__global__ __launch_bounds__(256) void scores_kernel(const unsigned short* __restrict__ qbf,
                                                     const unsigned short* __restrict__ kbf,
                                                     const int* __restrict__ eos,
                                                     float* __restrict__ scores) {
    __shared__ __align__(16) unsigned short ldsA[64 * kLdsStride];
    __shared__ __align__(16) unsigned short ldsB[128 * kLdsStride];

    const int tid  = threadIdx.x;
    const int wave = tid >> 5, lane = tid & 31;
    const int wm = wave >> 2, wn = wave & 3;
    const int half = lane >> 4, l15 = lane & 15;
    const int iBase = blockIdx.x * 64;
    const int jBase = blockIdx.y * 128;
    const int b     = blockIdx.z;

    const unsigned short* qb = qbf + (size_t)b * kS * kD;
    const unsigned short* kb = kbf + (size_t)b * kS * kD;

    GemmAcc acc = {};
    gemm_core(qb, kb, iBase, jBase, ldsA, ldsB, acc, wm, wn, half, l15);

#pragma unroll
    for (int ti = 0; ti < 2; ++ti)
#pragma unroll
        for (int tj = 0; tj < 2; ++tj)
#pragma unroll
            for (int r = 0; r < 8; ++r) {
                const int i = iBase + wm * 32 + ti * 16 + half * 8 + r;
                const int j = jBase + wn * 32 + tj * 16 + l15;
                float v = acc.a[ti][tj][r] * (1.0f / (float)kD);
                const size_t idx = (size_t)b * kSS + (size_t)i * kS + j;
                const int e = eos[idx];
                if (e != 0 && (j - i == 1 || i - j == 1)) v += 1e-19f;
                scores[idx] = v;
            }
}

// =====================================================================
// Kernel 5: row softmax -> neibor = prior + (1-prior)*(softmax + 1e-9)
// =====================================================================
__global__ __launch_bounds__(256) void softmax_kernel(float* __restrict__ sc,
                                                      const float* __restrict__ prior) {
    const size_t row = blockIdx.x;
    float* p = sc + row * (size_t)kS;
    const int tid = threadIdx.x;

    const float4 a = ((const float4*)p)[tid * 2];
    const float4 c = ((const float4*)p)[tid * 2 + 1];
    float r[8] = {a.x, a.y, a.z, a.w, c.x, c.y, c.z, c.w};

    __shared__ float red[256];
    float mx = r[0];
#pragma unroll
    for (int e = 1; e < 8; ++e) mx = fmaxf(mx, r[e]);
    red[tid] = mx; __syncthreads();
    for (int off = 128; off > 0; off >>= 1) {
        if (tid < off) red[tid] = fmaxf(red[tid], red[tid + off]);
        __syncthreads();
    }
    mx = red[0]; __syncthreads();

    float sm = 0.f;
#pragma unroll
    for (int e = 0; e < 8; ++e) { r[e] = expf(r[e] - mx); sm += r[e]; }
    red[tid] = sm; __syncthreads();
    for (int off = 128; off > 0; off >>= 1) {
        if (tid < off) red[tid] += red[tid + off];
        __syncthreads();
    }
    const float inv = 1.0f / red[0];
    const float pr  = prior[0];

    float4 oa, oc;
    oa.x = pr + (1.f - pr) * (r[0] * inv + 1e-9f);
    oa.y = pr + (1.f - pr) * (r[1] * inv + 1e-9f);
    oa.z = pr + (1.f - pr) * (r[2] * inv + 1e-9f);
    oa.w = pr + (1.f - pr) * (r[3] * inv + 1e-9f);
    oc.x = pr + (1.f - pr) * (r[4] * inv + 1e-9f);
    oc.y = pr + (1.f - pr) * (r[5] * inv + 1e-9f);
    oc.z = pr + (1.f - pr) * (r[6] * inv + 1e-9f);
    oc.w = pr + (1.f - pr) * (r[7] * inv + 1e-9f);
    ((float4*)p)[tid * 2]     = oa;
    ((float4*)p)[tid * 2 + 1] = oc;
}

// =====================================================================
// Kernel 6: per-batch exclusive prefix sum of log(na[i,i+1]+1e-9)
// =====================================================================
__global__ __launch_bounds__(256) void prefix_kernel(const float* __restrict__ na,
                                                     float* __restrict__ c) {
    const int b = blockIdx.x;
    const int tid = threadIdx.x;
    const float* nb = na + (size_t)b * kSS;

    float v[8];
    float tot = 0.f;
    for (int e = 0; e < 8; ++e) {
        const int i = tid * 8 + e;
        const float val = (i < kS - 1) ? logf(nb[(size_t)i * kS + i + 1] + 1e-9f) : 0.f;
        v[e] = tot;
        tot += val;
    }
    __shared__ float sh[256];
    sh[tid] = tot; __syncthreads();
    if (tid == 0) {
        float run = 0.f;
        for (int t = 0; t < 256; ++t) { const float x = sh[t]; sh[t] = run; run += x; }
    }
    __syncthreads();
    const float base = sh[tid];
    for (int e = 0; e < 8; ++e) c[(size_t)b * kS + tid * 8 + e] = base + v[e];
}

// =====================================================================
// Kernel 7: g_attn fill
// =====================================================================
__global__ __launch_bounds__(256) void gattn_kernel(const float* __restrict__ na,
                                                    const float* __restrict__ c,
                                                    float* __restrict__ g) {
    const unsigned idx = blockIdx.x * 256u + threadIdx.x;   // < 2^24
    const unsigned b   = idx >> 22;
    const unsigned rem = idx & 4194303u;
    const unsigned i   = rem >> 11;
    const unsigned j   = rem & 2047u;
    float out;
    if (i == j) {
        out = na[idx] + 1e-9f;
    } else {
        const unsigned lo = i < j ? i : j;
        const unsigned hi = i < j ? j : i;
        const float* cb = c + ((size_t)b << 11);
        out = expf(cb[hi] - cb[lo]) + 1e-9f;
    }
    g[idx] = out;
}

// =====================================================================
// launch
// =====================================================================
extern "C" void kernel_launch(void* const* d_in, const int* in_sizes, int n_in,
                              void* d_out, int out_size, void* d_ws, size_t ws_size,
                              hipStream_t stream) {
    (void)in_sizes; (void)n_in; (void)out_size; (void)ws_size;

    const float* ctx   = (const float*)d_in[0];
    const int*   eos   = (const int*)  d_in[1];
    const float* prior = (const float*)d_in[2];
    const float* gamma = (const float*)d_in[3];
    const float* beta  = (const float*)d_in[4];
    const float* Wk    = (const float*)d_in[5];
    const float* bk    = (const float*)d_in[6];
    const float* Wq    = (const float*)d_in[7];
    const float* bq    = (const float*)d_in[8];

    float* g_out = (float*)d_out;                      // g_attn   [B,S,S]
    float* neib  = g_out + (size_t)kB * kSS;           // neibor   [B,S,S]

    char* ws = (char*)d_ws;
    unsigned short* xbf  = (unsigned short*)(ws);                 // 16 MB
    unsigned short* wqt  = (unsigned short*)(ws + 16777216);      //  2 MB (transposed)
    unsigned short* wkt  = (unsigned short*)(ws + 18874368);      //  2 MB (transposed)
    unsigned short* qbf  = (unsigned short*)(ws + 20971520);      // 16 MB
    unsigned short* kbf  = (unsigned short*)(ws + 37748736);      // 16 MB
    float*          cpre = (float*)        (ws + 54525952);       // 32 KB

    ln_kernel<<<kB * kS, 256, 0, stream>>>(ctx, gamma, beta, xbf);

    dim3 gt(kD / 64, kD / 64);                          // 16 x 16
    transposebf_kernel<<<gt, 256, 0, stream>>>(Wq, wqt);
    transposebf_kernel<<<gt, 256, 0, stream>>>(Wk, wkt);

    dim3 g1(kB * kS / 64, kD / 128);                    // 128 x 8
    gemm_proj_kernel<<<g1, 256, 0, stream>>>(xbf, wqt, bq, qbf);
    gemm_proj_kernel<<<g1, 256, 0, stream>>>(xbf, wkt, bk, kbf);

    dim3 g2(kS / 64, kS / 128, kB);                     // 32 x 16 x 4
    scores_kernel<<<g2, 256, 0, stream>>>(qbf, kbf, eos, neib);

    softmax_kernel<<<kB * kS, 256, 0, stream>>>(neib, prior);
    prefix_kernel<<<kB, 256, 0, stream>>>(neib, cpre);
    gattn_kernel<<<(unsigned)((size_t)kB * kSS / 256), 256, 0, stream>>>(neib, cpre, g_out);
}
